// Ensemble_65017214927503
// MI455X (gfx1250) — matrix operand order repaired and checked by reading, not compiled, not run on hardware
//
#include <hip/hip_runtime.h>

// ---------------------------------------------------------------------------
// Problem constants (from the reference)
// ---------------------------------------------------------------------------
constexpr int B_TOT  = 262144;
constexpr int IN_DIM = 64;
constexpr int HID    = 256;
constexpr int OUT_D  = 16;
constexpr int NHEAD  = 8;
constexpr int OH     = OUT_D * NHEAD;          // 128
constexpr float INIT_SCALE_T = -0.4327521295671885f; // log(expm1(0.5))
constexpr float MIN_SCALE = 1e-6f;
constexpr float MAX_SCALE = 1.0f;

typedef __attribute__((ext_vector_type(16))) __bf16 v16bf;
typedef __attribute__((ext_vector_type(8)))  __bf16 v8bf;
typedef __attribute__((ext_vector_type(8)))  float  v8f;

__device__ __forceinline__ v8f wmma_bf16(v16bf a, v16bf b, v8f c) {
    // D = A(16x32 bf16) * B(32x16 bf16) + C(16x16 f32)
    return __builtin_amdgcn_wmma_f32_16x16x32_bf16(
        /*neg_a=*/false, a, /*neg_b=*/false, b,
        /*c_mod=*/(short)0, c, /*reuse_a=*/false, /*reuse_b=*/false);
}

// CDNA5 has a hardware transcendental TANH (ISA §8.4 TRANS list).
__device__ __forceinline__ float fast_tanh(float x) {
#if __has_builtin(__builtin_amdgcn_tanhf)
    return __builtin_amdgcn_tanhf(x);
#elif __has_builtin(__builtin_amdgcn_tanh_f32)
    return __builtin_amdgcn_tanh_f32(x);
#else
    float r;
    // v_nop satisfies the TRANS-op result-use hazard (compiler can't see it).
    asm volatile("v_tanh_f32 %0, %1\n\tv_nop" : "=v"(r) : "v"(x));
    return r;
#endif
}

__device__ __forceinline__ float softplus_f(float x) {
    // stable softplus with native exp/log (args to log are in (1,2])
    float e = __expf(-fabsf(x));
    return fmaxf(x, 0.0f) + __logf(1.0f + e);
}

// ---------------------------------------------------------------------------
// Pack W[K][N] (f32 row-major) into WMMA-B-friendly bf16 layout:
//   dst[((ks*N)+n)*16 + j] = bf16( W[((ks>>1)*32 + (ks&1)*8 + (j<8 ? j : j+8))*N + n] ),  ks = 0..K/16-1
// ---------------------------------------------------------------------------
__global__ void pack_w_kernel(const float* __restrict__ W, __bf16* __restrict__ dst,
                              int K, int N) {
    int idx = blockIdx.x * blockDim.x + threadIdx.x;   // over (K/16)*N
    int total = (K / 16) * N;
    if (idx >= total) return;
    int ks = idx / N;
    int n  = idx - ks * N;
    __bf16* d = dst + (size_t)idx * 16;
    #pragma unroll
    for (int j = 0; j < 16; ++j)
        d[j] = (__bf16)W[(size_t)((ks >> 1) * 32 + (ks & 1) * 8 + (j < 8 ? j : j + 8)) * N + n];
}

// ---------------------------------------------------------------------------
// Zero-fill scale_tril region (bulk of output bytes) at full bandwidth.
// ---------------------------------------------------------------------------
__global__ void zero_f4_kernel(float4* __restrict__ p, int n4) {
    int i = blockIdx.x * blockDim.x + threadIdx.x;
    if (i < n4) p[i] = float4{0.f, 0.f, 0.f, 0.f};
}

// ---------------------------------------------------------------------------
// Fused MLP: each wave32 owns a 16-row tile. 8 waves / 256-thread workgroup.
// LDS: [0,128K)    packed bf16 w2 (staged once per workgroup, async if avail)
//      [128K,192K) per-wave 8KB buffer: bf16 h[16][256]  (aliased f32[16][128])
// ---------------------------------------------------------------------------
__global__ __launch_bounds__(256, 1) void mlp_fused(
    const float* __restrict__ x,
    const float* __restrict__ b1, const float* __restrict__ b2,
    const float* __restrict__ bh, const float* __restrict__ bs,
    const int*   __restrict__ head_idx,
    const __bf16* __restrict__ w1p, const __bf16* __restrict__ w2p,
    const __bf16* __restrict__ whp, const __bf16* __restrict__ wsp,
    float* __restrict__ out_mean, float* __restrict__ out_tril)
{
    extern __shared__ char smem[];
    __bf16* w2s = (__bf16*)smem;                       // 65536 bf16 = 128 KB

    const int tid  = threadIdx.x;
    const int wave = tid >> 5;
    const int lane = tid & 31;

    // ---- stage packed w2 into LDS (128 KB) ----
    {
#if __has_builtin(__builtin_amdgcn_global_load_async_to_lds_b128) && \
    __has_builtin(__builtin_amdgcn_s_wait_asynccnt)
        typedef int v4i __attribute__((__vector_size__(16)));
        typedef __attribute__((address_space(1))) v4i gv4i;
        typedef __attribute__((address_space(3))) v4i lv4i;
        gv4i* src = (gv4i*)w2p;
        lv4i* dst = (lv4i*)smem;
        // 131072 B / 256 threads = 512 B per thread = 32 x b128 async copies
        #pragma unroll
        for (int i = 0; i < 32; ++i) {
            int idx = tid + i * 256;
            __builtin_amdgcn_global_load_async_to_lds_b128(src + idx, dst + idx, 0, 0);
        }
        __builtin_amdgcn_s_wait_asynccnt(0);
#else
        const uint4* src = (const uint4*)w2p;
        uint4*       dst = (uint4*)smem;
        #pragma unroll 4
        for (int i = tid; i < (HID * HID) / 8; i += 256) dst[i] = src[i];
#endif
    }
    __syncthreads();

    __bf16* hrow = (__bf16*)(smem + 131072 + wave * 8192);  // bf16 [16][256]
    float*  fbuf = (float*)hrow;                            // f32  [16][128] alias

    const int gwave = blockIdx.x * 8 + wave;
    const int row0  = gwave * 16;
    const int m     = lane & 15;   // row (A) or col (B/C)
    const int hh    = lane >> 4;   // lane half

    // ---- A1 fragments from x (f32 -> bf16), ISA A-layout:
    // slots 0..7 : K = base + 8*hh + e ; slots 8..15 : K = base + 16 + 8*hh + e
    v16bf a1[2];
    #pragma unroll
    for (int ks = 0; ks < 2; ++ks) {
        const float* p = x + (size_t)(row0 + m) * IN_DIM + ks * 32 + hh * 8;
        #pragma unroll
        for (int e = 0; e < 8; ++e) {
            a1[ks][e]     = (__bf16)p[e];
            a1[ks][e + 8] = (__bf16)p[e + 16];
        }
    }

    // ---- layer 1: h1 = tanh(x*w1 + b1) -> LDS (bf16) ----
    for (int nt = 0; nt < 16; ++nt) {
        v8f acc = {0.f, 0.f, 0.f, 0.f, 0.f, 0.f, 0.f, 0.f};
        #pragma unroll
        for (int ks = 0; ks < 2; ++ks) {
            const __bf16* pb = w1p + ((size_t)(ks * 2 + hh) * HID + nt * 16 + m) * 16;
            acc = wmma_bf16(a1[ks], *(const v16bf*)pb, acc);
        }
        float bias = b1[nt * 16 + m];
        #pragma unroll
        for (int r = 0; r < 8; ++r) {
            float v = fast_tanh(acc[r] + bias);
            hrow[(r + 8 * hh) * HID + nt * 16 + m] = (__bf16)v;
        }
    }

    // ---- load A2 fragments (all K=256) from LDS into registers ----
    v16bf af[8];
    #pragma unroll
    for (int ks = 0; ks < 8; ++ks) {
        v8bf c0 = *(const v8bf*)(hrow + m * HID + ks * 32 + hh * 8);
        v8bf c1 = *(const v8bf*)(hrow + m * HID + ks * 32 + 16 + hh * 8);
        #pragma unroll
        for (int e = 0; e < 8; ++e) { af[ks][e] = c0[e]; af[ks][e + 8] = c1[e]; }
    }

    // ---- layer 2: h2 = tanh(h1*w2 + b2), B-fragments from LDS ----
    for (int nt = 0; nt < 16; ++nt) {
        v8f acc = {0.f, 0.f, 0.f, 0.f, 0.f, 0.f, 0.f, 0.f};
        #pragma unroll
        for (int ks = 0; ks < 8; ++ks) {
            const __bf16* pb = w2s + ((ks * 2 + hh) * HID + nt * 16 + m) * 16;
            acc = wmma_bf16(af[ks], *(const v16bf*)pb, acc);
        }
        float bias = b2[nt * 16 + m];
        #pragma unroll
        for (int r = 0; r < 8; ++r) {
            float v = fast_tanh(acc[r] + bias);
            hrow[(r + 8 * hh) * HID + nt * 16 + m] = (__bf16)v;
        }
    }

    // ---- reload A fragments with h2 ----
    #pragma unroll
    for (int ks = 0; ks < 8; ++ks) {
        v8bf c0 = *(const v8bf*)(hrow + m * HID + ks * 32 + hh * 8);
        v8bf c1 = *(const v8bf*)(hrow + m * HID + ks * 32 + 16 + hh * 8);
        #pragma unroll
        for (int e = 0; e < 8; ++e) { af[ks][e] = c0[e]; af[ks][e + 8] = c1[e]; }
    }

    const int hid = head_idx[row0 + m];   // per-row head selection

    // ---- mean head: out = h2*wh + bh -> f32 LDS, then gather ----
    for (int nt = 0; nt < 8; ++nt) {
        v8f acc = {0.f, 0.f, 0.f, 0.f, 0.f, 0.f, 0.f, 0.f};
        #pragma unroll
        for (int ks = 0; ks < 8; ++ks) {
            const __bf16* pb = whp + ((size_t)(ks * 2 + hh) * OH + nt * 16 + m) * 16;
            acc = wmma_bf16(af[ks], *(const v16bf*)pb, acc);
        }
        float bias = bh[nt * 16 + m];
        #pragma unroll
        for (int r = 0; r < 8; ++r)
            fbuf[(r + 8 * hh) * OH + nt * 16 + m] = acc[r] + bias;
    }
    // gather mean[b, d] = out[b, d*8 + head]; lane (m,hh) covers d = 8*hh..8*hh+7
    #pragma unroll
    for (int d8 = 0; d8 < 8; ++d8) {
        int d = hh * 8 + d8;
        out_mean[(size_t)(row0 + m) * OUT_D + d] = fbuf[m * OH + d * 8 + hid];
    }

    // ---- scale head: s = h2*ws + bs + init, double-softplus clamp ----
    for (int nt = 0; nt < 8; ++nt) {
        v8f acc = {0.f, 0.f, 0.f, 0.f, 0.f, 0.f, 0.f, 0.f};
        #pragma unroll
        for (int ks = 0; ks < 8; ++ks) {
            const __bf16* pb = wsp + ((size_t)(ks * 2 + hh) * OH + nt * 16 + m) * 16;
            acc = wmma_bf16(af[ks], *(const v16bf*)pb, acc);
        }
        float bias = bs[nt * 16 + m] + INIT_SCALE_T;
        #pragma unroll
        for (int r = 0; r < 8; ++r) {
            float s = acc[r] + bias;
            s = MAX_SCALE - softplus_f(MAX_SCALE - s);
            s = MIN_SCALE + softplus_f(s - MIN_SCALE);
            fbuf[(r + 8 * hh) * OH + nt * 16 + m] = s;
        }
    }
    // gather selected scale onto the diagonal of scale_tril[b] (16x16)
    #pragma unroll
    for (int d8 = 0; d8 < 8; ++d8) {
        int d = hh * 8 + d8;
        float sv = fbuf[m * OH + d * 8 + hid];
        out_tril[(size_t)(row0 + m) * (OUT_D * OUT_D) + d * (OUT_D + 1)] = sv;
    }
}

// ---------------------------------------------------------------------------
extern "C" void kernel_launch(void* const* d_in, const int* in_sizes, int n_in,
                              void* d_out, int out_size, void* d_ws, size_t ws_size,
                              hipStream_t stream) {
    const float* x  = (const float*)d_in[0];
    const float* w1 = (const float*)d_in[1];
    const float* b1 = (const float*)d_in[2];
    const float* w2 = (const float*)d_in[3];
    const float* b2 = (const float*)d_in[4];
    const float* wh = (const float*)d_in[5];
    const float* bh = (const float*)d_in[6];
    const float* ws = (const float*)d_in[7];
    const float* bs = (const float*)d_in[8];
    const int* hidx = (const int*)d_in[9];

    // packed bf16 weights in workspace (total ~288 KB)
    __bf16* w1p = (__bf16*)d_ws;                   // 4  * 256 * 16
    __bf16* w2p = w1p + (IN_DIM / 16) * HID * 16;  // 16 * 256 * 16
    __bf16* whp = w2p + (HID / 16) * HID * 16;     // 16 * 128 * 16
    __bf16* wsp = whp + (HID / 16) * OH * 16;      // 16 * 128 * 16

    pack_w_kernel<<<((IN_DIM / 16) * HID + 255) / 256, 256, 0, stream>>>(w1, w1p, IN_DIM, HID);
    pack_w_kernel<<<((HID / 16) * HID + 255) / 256, 256, 0, stream>>>(w2, w2p, HID, HID);
    pack_w_kernel<<<((HID / 16) * OH + 255) / 256, 256, 0, stream>>>(wh, whp, HID, OH);
    pack_w_kernel<<<((HID / 16) * OH + 255) / 256, 256, 0, stream>>>(ws, wsp, HID, OH);

    float* out_mean = (float*)d_out;
    float* out_tril = out_mean + (size_t)B_TOT * OUT_D;

    int n4 = B_TOT * OUT_D * OUT_D / 4;
    zero_f4_kernel<<<(n4 + 255) / 256, 256, 0, stream>>>((float4*)out_tril, n4);

    constexpr int SMEM = 131072 + 8 * 8192;  // w2 stage + 8 per-wave buffers
    (void)hipFuncSetAttribute((const void*)mlp_fused,
                              hipFuncAttributeMaxDynamicSharedMemorySize, SMEM);
    // 16384 wave-tiles / 8 waves per block = 2048 blocks (exact, no tail)
    mlp_fused<<<B_TOT / 16 / 8, 256, SMEM, stream>>>(
        x, b1, b2, bh, bs, hidx, w1p, w2p, whp, wsp, out_mean, out_tril);
}
